// PGPEncoder_5729486373130
// MI455X (gfx1250) — matrix operand order repaired and checked by reading, not compiled, not run on hardware
//
#include <hip/hip_runtime.h>
#include <hip/hip_bf16.h>
#include <math.h>

// ---------------------------------------------------------------------------
// PGP encoder for MI455X (gfx1250, wave32, WMMA).
// All heavy GEMMs use v_wmma_f32_16x16x32_bf16 (bf16 in, f32 accumulate).
// Attention probs@V stages V tiles into LDS with global_load_async_to_lds_b128
// (ASYNCcnt-tracked) before consuming them with WMMA.
// ---------------------------------------------------------------------------

typedef __attribute__((ext_vector_type(16))) __bf16 v16bf;
typedef __attribute__((ext_vector_type(8)))  __bf16 v8bf;
typedef __attribute__((ext_vector_type(8)))  float  v8f;

#define LRELU(x) ((x) > 0.f ? (x) : 0.01f * (x))

// ---------------------------------------------------------------------------
// f32 -> bf16 convert (parameters, one launch per tensor)
// ---------------------------------------------------------------------------
__global__ void f32_to_bf16_k(const float* __restrict__ s, __bf16* __restrict__ d, int n) {
    int i = blockIdx.x * blockDim.x + threadIdx.x;
    if (i < n) d[i] = (__bf16)s[i];
}

// ---------------------------------------------------------------------------
// Embedding: out = lrelu(X @ W^T + b), optional constant class-flag feature,
// also emits per-(seq,t) validity byte from the float mask convention.
// ---------------------------------------------------------------------------
__global__ void embed_k(const float* __restrict__ X, const float* __restrict__ mask,
                        const float* __restrict__ W, const float* __restrict__ bias,
                        __bf16* __restrict__ out, unsigned char* __restrict__ valid,
                        int Bb, int A_in, int A_tot, int a_off, int T, int F, int Fin,
                        float flagVal) {
    int total = Bb * A_in * T * 64;
    for (int idx = blockIdx.x * blockDim.x + threadIdx.x; idx < total;
         idx += gridDim.x * blockDim.x) {
        int unit = idx & 63;
        int row  = idx >> 6;
        int b = row / (A_in * T);
        int rem = row - b * A_in * T;
        int a = rem / T;
        int t = rem - a * T;
        const float* x = X + (size_t)row * F;
        const float* w = W + (size_t)unit * Fin;
        float acc = bias[unit];
        for (int f = 0; f < F; ++f) acc += x[f] * w[f];
        if (Fin > F) acc += flagVal * w[F];
        float y = LRELU(acc);
        int seq = b * A_tot + a_off + a;
        out[((size_t)seq * T + t) * 64 + unit] = (__bf16)y;
        if (unit == 0) {
            unsigned char v = mask ? (unsigned char)(mask[(size_t)row * F] == 0.0f)
                                   : (unsigned char)1;
            valid[(size_t)seq * T + t] = v;
        }
    }
}

// ---------------------------------------------------------------------------
// WMMA fragment packing per CDNA5 ISA layouts.
// A (16x32 bf16): lane L: M=L&15, g=L>>4; element i -> K=(i&7)+8g+16*(i>>3)
// B (32x16 bf16): lane L: N=L&15, g=L>>4; element i -> K=(i&7)+8g+16*(i>>3)
// C/D (16x16 f32, v8f): lane L vgpr v -> row v+8g, col L&15.
// ---------------------------------------------------------------------------
__device__ inline v16bf load_a_frag(const __bf16* rowPtr, int g) {
    v8bf lo = *(const v8bf*)(rowPtr + 8 * g);
    v8bf hi = *(const v8bf*)(rowPtr + 16 + 8 * g);
    v16bf a;
#pragma unroll
    for (int i = 0; i < 8; ++i) { a[i] = lo[i]; a[i + 8] = hi[i]; }
    return a;
}
__device__ inline v16bf load_b_frag(const __bf16* rowPtr, int g) {
    v8bf w0 = *(const v8bf*)(rowPtr + 8 * g);
    v8bf w1 = *(const v8bf*)(rowPtr + 16 + 8 * g);
    v16bf b;
#pragma unroll
    for (int i = 0; i < 8; ++i) { b[i] = w0[i]; b[i + 8] = w1[i]; }
    return b;
}

// ---------------------------------------------------------------------------
// Generic GEMM: C[M,N] = act(A[M,K] @ W[N,K]^T + bias), A,W bf16, C f32,
// optional bf16 mirror. One wave per block, 16(M) x 64(N) tile.
// grid = (ceil(M/16), N/64), block = 32.
// ---------------------------------------------------------------------------
__global__ void wmma_gemm_k(const __bf16* __restrict__ A, const __bf16* __restrict__ W,
                            const float* __restrict__ bias, float* __restrict__ C,
                            __bf16* __restrict__ Cbf, int M, int N, int K, int act) {
    int lane = threadIdx.x;
    int g = lane >> 4, ml = lane & 15;
    int mt = blockIdx.x;
    int nt0 = blockIdx.y * 4;
    int row = mt * 16 + ml;
    int arow = row < M ? row : M - 1;
    v8f acc0 = {}, acc1 = {}, acc2 = {}, acc3 = {};
    int KT = K / 32;
    for (int kt = 0; kt < KT; ++kt) {
        v16bf a = load_a_frag(A + (size_t)arow * K + kt * 32, g);
        if (kt + 1 < KT)  // gfx1250 global_prefetch_b8 of next A tile
            __builtin_prefetch(A + (size_t)arow * K + (kt + 1) * 32, 0, 1);
        {
            v16bf b = load_b_frag(W + (size_t)((nt0 + 0) * 16 + ml) * K + kt * 32, g);
            acc0 = __builtin_amdgcn_wmma_f32_16x16x32_bf16(false, a, false, b, (short)0, acc0, false, false);
        }
        {
            v16bf b = load_b_frag(W + (size_t)((nt0 + 1) * 16 + ml) * K + kt * 32, g);
            acc1 = __builtin_amdgcn_wmma_f32_16x16x32_bf16(false, a, false, b, (short)0, acc1, false, false);
        }
        {
            v16bf b = load_b_frag(W + (size_t)((nt0 + 2) * 16 + ml) * K + kt * 32, g);
            acc2 = __builtin_amdgcn_wmma_f32_16x16x32_bf16(false, a, false, b, (short)0, acc2, false, false);
        }
        {
            v16bf b = load_b_frag(W + (size_t)((nt0 + 3) * 16 + ml) * K + kt * 32, g);
            acc3 = __builtin_amdgcn_wmma_f32_16x16x32_bf16(false, a, false, b, (short)0, acc3, false, false);
        }
    }
    v8f* accs[4] = { &acc0, &acc1, &acc2, &acc3 };
#pragma unroll
    for (int n = 0; n < 4; ++n) {
        int col = (nt0 + n) * 16 + ml;
        float bv = bias ? bias[col] : 0.f;
#pragma unroll
        for (int v = 0; v < 8; ++v) {
            int r = mt * 16 + v + 8 * g;
            if (r < M) {
                float x = (*accs[n])[v] + bv;
                if (act == 1) x = LRELU(x);
                size_t o = (size_t)r * N + col;
                C[o] = x;
                if (Cbf) Cbf[o] = (__bf16)x;
            }
        }
    }
}

// ---------------------------------------------------------------------------
// Fused GRU: one wave handles 16 sequences, T steps, E=64 input, H=128 hidden.
// Per step: Gi = emb_t @ Wih^T + bih, Gh = h @ Whh^T + bhh via WMMA (N=384),
// then elementwise gates; hidden frozen on invalid (padded) steps.
// Whh/Wih stay in L2 (147 KB shared by all 1024 waves; MI455X L2 = 192 MB).
// grid = (M/16), block = 32.
// ---------------------------------------------------------------------------
__global__ void gru_fused_k(const __bf16* __restrict__ emb /*[M,T,64]*/,
                            const unsigned char* __restrict__ valid /*[M,T]*/,
                            const __bf16* __restrict__ Wih /*[384,64]*/,
                            const __bf16* __restrict__ Whh /*[384,128]*/,
                            const float* __restrict__ bih, const float* __restrict__ bhh,
                            float* __restrict__ Hout /*[M,128]*/,
                            __bf16* __restrict__ HoutBf, int M, int T) {
    __shared__ float hbuf[16][128];
    __shared__ float Gi[16][384];
    __shared__ float Gh[16][384];
    int lane = threadIdx.x;
    int g = lane >> 4, ml = lane & 15;
    int seq0 = blockIdx.x * 16;
    int srow = seq0 + ml;
    if (srow >= M) srow = M - 1;
    for (int e = lane; e < 16 * 128; e += 32) hbuf[e >> 7][e & 127] = 0.f;
    __syncthreads();

    for (int t = 0; t < T; ++t) {
        v16bf ae0 = load_a_frag(emb + ((size_t)srow * T + t) * 64 + 0, g);
        v16bf ae1 = load_a_frag(emb + ((size_t)srow * T + t) * 64 + 32, g);
        v16bf ah[4];
#pragma unroll
        for (int kt = 0; kt < 4; ++kt) {
#pragma unroll
            for (int i = 0; i < 16; ++i) {
                int K = (i & 7) + 8 * g + 16 * (i >> 3);
                ah[kt][i] = (__bf16)hbuf[ml][kt * 32 + K];
            }
        }
        for (int nt = 0; nt < 24; ++nt) {
            int ncol = nt * 16 + ml;
            v8f accH = {};
#pragma unroll
            for (int kt = 0; kt < 4; ++kt) {
                v16bf b = load_b_frag(Whh + (size_t)ncol * 128 + kt * 32, g);
                accH = __builtin_amdgcn_wmma_f32_16x16x32_bf16(false, ah[kt], false, b, (short)0, accH, false, false);
            }
            v8f accI = {};
            {
                v16bf b = load_b_frag(Wih + (size_t)ncol * 64 + 0, g);
                accI = __builtin_amdgcn_wmma_f32_16x16x32_bf16(false, ae0, false, b, (short)0, accI, false, false);
            }
            {
                v16bf b = load_b_frag(Wih + (size_t)ncol * 64 + 32, g);
                accI = __builtin_amdgcn_wmma_f32_16x16x32_bf16(false, ae1, false, b, (short)0, accI, false, false);
            }
            float bh = bhh[ncol], bi = bih[ncol];
#pragma unroll
            for (int v = 0; v < 8; ++v) {
                int r = v + 8 * g;
                Gh[r][ncol] = accH[v] + bh;
                Gi[r][ncol] = accI[v] + bi;
            }
        }
        __syncthreads();
        for (int e = lane; e < 16 * 128; e += 32) {
            int s = e >> 7, u = e & 127;
            float ir = Gi[s][u], iz = Gi[s][u + 128], inn = Gi[s][u + 256];
            float hr = Gh[s][u], hz = Gh[s][u + 128], hn = Gh[s][u + 256];
            float r = 1.f / (1.f + __expf(-(ir + hr)));
            float z = 1.f / (1.f + __expf(-(iz + hz)));
            float n = tanhf(inn + r * hn);
            float hold = hbuf[s][u];
            float hnew = (1.f - z) * n + z * hold;
            int sr = seq0 + s;
            unsigned char vld = (sr < M) ? valid[(size_t)sr * T + t] : (unsigned char)0;
            hbuf[s][u] = vld ? hnew : hold;
        }
        __syncthreads();
    }
    for (int e = lane; e < 16 * 128; e += 32) {
        int s = e >> 7, u = e & 127;
        int sr = seq0 + s;
        if (sr < M) {
            float x = hbuf[s][u];
            Hout[(size_t)sr * 128 + u] = x;
            if (HoutBf) HoutBf[(size_t)sr * 128 + u] = (__bf16)x;
        }
    }
}

// ---------------------------------------------------------------------------
// Single-head attention with bool mask (1 = disallowed), d = 128.
// One wave per 16-query tile: scores (WMMA) -> masked softmax (LDS) ->
// context: V tiles async-staged into LDS (global_load_async_to_lds_b128,
// ASYNCcnt), then consumed by WMMA. grid = (Lq/16, B), block = 32.
// ---------------------------------------------------------------------------
__global__ void attn_k(const __bf16* __restrict__ qh, const __bf16* __restrict__ kh,
                       const __bf16* __restrict__ vh, const unsigned char* __restrict__ mask,
                       float* __restrict__ out, __bf16* __restrict__ outbf,
                       int Lq, int S, float scale) {
    __shared__ float  sc[16][512];
    __shared__ __bf16 pb[16][512];
    __shared__ __bf16 vtile[32][128];   // 8 KB async staging buffer
    int b = blockIdx.y;
    int l0 = blockIdx.x * 16;
    int lane = threadIdx.x;
    int g = lane >> 4, ml = lane & 15;

    // ---- phase 1: scores = qh @ kh^T (WMMA over d=128) ----
    const __bf16* qbase = qh + ((size_t)b * Lq + l0) * 128;
    v16bf aq[4];
#pragma unroll
    for (int kt = 0; kt < 4; ++kt)
        aq[kt] = load_a_frag(qbase + (size_t)ml * 128 + kt * 32, g);

    const __bf16* kbase = kh + (size_t)b * S * 128;
    for (int st = 0; st < S / 16; ++st) {
        v8f acc = {};
        int srow = st * 16 + ml;
#pragma unroll
        for (int kt = 0; kt < 4; ++kt) {
            v16bf bb = load_b_frag(kbase + (size_t)srow * 128 + kt * 32, g);
            acc = __builtin_amdgcn_wmma_f32_16x16x32_bf16(false, aq[kt], false, bb, (short)0, acc, false, false);
        }
#pragma unroll
        for (int v = 0; v < 8; ++v) {
            int r = v + 8 * g;
            int col = st * 16 + ml;
            unsigned char mk = mask[((size_t)b * Lq + (l0 + r)) * S + col];
            sc[r][col] = mk ? -1e30f : acc[v] * scale;
        }
    }
    __syncthreads();
    // ---- phase 2: masked softmax (rows in LDS) ----
    if (lane < 16) {
        int r = lane;
        float mx = -1e30f;
        for (int s2 = 0; s2 < S; ++s2) mx = fmaxf(mx, sc[r][s2]);
        float sum = 0.f;
        for (int s2 = 0; s2 < S; ++s2) { float e = __expf(sc[r][s2] - mx); sc[r][s2] = e; sum += e; }
        float inv = 1.f / sum;
        for (int s2 = 0; s2 < S; ++s2) pb[r][s2] = (__bf16)(sc[r][s2] * inv);
    }
    __syncthreads();
    // ---- phase 3: out = probs @ V, V async-staged into LDS per 32-key tile ----
    const __bf16* vbase = vh + (size_t)b * S * 128;
    unsigned vtbase = (unsigned)(size_t)(&vtile[0][0]);  // LDS byte offset
    v8f acc[8];
#pragma unroll
    for (int dt = 0; dt < 8; ++dt) acc[dt] = (v8f){};

    for (int kt = 0; kt < S / 32; ++kt) {
        const char* gsrc = (const char*)(vbase + (size_t)kt * 32 * 128);
        // 8 KB tile: 16 wave-wide async b128 transfers (512 B each)
#pragma unroll
        for (int c = 0; c < 16; ++c) {
            unsigned loff = vtbase + (unsigned)(c * 512 + lane * 16);
            unsigned long long ga =
                (unsigned long long)gsrc + (unsigned long long)(c * 512 + lane * 16);
            asm volatile("global_load_async_to_lds_b128 %0, %1, off"
                         :: "v"(loff), "v"(ga) : "memory");
        }
        asm volatile("s_wait_asynccnt 0x0" ::: "memory");
        __syncthreads();
        // A-fragment (probs) depends only on kt: build once
        v16bf a;
#pragma unroll
        for (int i = 0; i < 16; ++i) {
            int K = (i & 7) + 8 * g + 16 * (i >> 3);
            a[i] = pb[ml][kt * 32 + K];
        }
#pragma unroll
        for (int dt = 0; dt < 8; ++dt) {
            v16bf bb;
#pragma unroll
            for (int i = 0; i < 16; ++i)
                bb[i] = vtile[(i & 7) + 8 * g + 16 * (i >> 3)][dt * 16 + ml];
            acc[dt] = __builtin_amdgcn_wmma_f32_16x16x32_bf16(false, a, false, bb, (short)0, acc[dt], false, false);
        }
        __syncthreads();   // all lanes done with vtile before next async overwrite
    }
#pragma unroll
    for (int dt = 0; dt < 8; ++dt) {
#pragma unroll
        for (int v = 0; v < 8; ++v) {
            int r = v + 8 * g;
            int col = dt * 16 + ml;
            size_t o = ((size_t)b * Lq + l0 + r) * 128 + col;
            out[o] = acc[dt][v];
            if (outbf) outbf[o] = (__bf16)acc[dt][v];
        }
    }
}

// ---------------------------------------------------------------------------
// Small elementwise helpers
// ---------------------------------------------------------------------------
__global__ void build_attmask_k(const unsigned char* __restrict__ vnm,
                                const unsigned char* __restrict__ pnm,
                                unsigned char* __restrict__ out, int total) {
    for (int i = blockIdx.x * blockDim.x + threadIdx.x; i < total; i += gridDim.x * blockDim.x) {
        int s = i % 160;
        int bn = i / 160;
        out[i] = (s < 96) ? vnm[(size_t)bn * 96 + s] : pnm[(size_t)bn * 64 + (s - 96)];
    }
}

__global__ void build_adjmask_k(const int* __restrict__ s_next, const int* __restrict__ et,
                                unsigned char* __restrict__ mask, int Nn, int NE) {
    int bi = blockIdx.x;
    unsigned char* row = mask + (size_t)bi * Nn;
    for (int j = threadIdx.x; j < Nn; j += blockDim.x) row[j] = 1;
    __syncthreads();
    if (threadIdx.x == 0) {
        int i = bi % Nn;
        row[i] = 0;
        const int* sp = s_next + (size_t)bi * NE;
        const int* ep = et + (size_t)bi * NE;
        for (int k2 = 0; k2 < NE; ++k2) {
            int tgt = (ep[k2] > 0) ? sp[k2] : i;
            if (tgt >= 0 && tgt < Nn) row[tgt] = 0;
        }
    }
}

__global__ void concat2_bf_k(const __bf16* __restrict__ a, const __bf16* __restrict__ b,
                             __bf16* __restrict__ o, int Mrows) {
    int total = Mrows * 256;
    for (int i = blockIdx.x * blockDim.x + threadIdx.x; i < total; i += gridDim.x * blockDim.x) {
        int r = i >> 8, c = i & 255;
        o[i] = (c < 128) ? a[(size_t)r * 128 + c] : b[(size_t)r * 128 + (c - 128)];
    }
}

__global__ void add_resid_k(float* __restrict__ ne, const float* __restrict__ d,
                            __bf16* __restrict__ nebf, int n) {
    for (int i = blockIdx.x * blockDim.x + threadIdx.x; i < n; i += gridDim.x * blockDim.x) {
        float x = ne[i] + d[i];
        ne[i] = x;
        nebf[i] = (__bf16)x;
    }
}

__global__ void copy_f32_k(const float* __restrict__ s, float* __restrict__ d, int n) {
    for (int i = blockIdx.x * blockDim.x + threadIdx.x; i < n; i += gridDim.x * blockDim.x)
        d[i] = s[i];
}

__global__ void combined_mask_k(const unsigned char* __restrict__ valid, float* __restrict__ out,
                                int M, int T) {
    for (int i = blockIdx.x * blockDim.x + threadIdx.x; i < M; i += gridDim.x * blockDim.x) {
        unsigned char any = 0;
        for (int t = 0; t < T; ++t) any |= valid[(size_t)i * T + t];
        out[i] = any ? 0.f : 1.f;
    }
}

// ---------------------------------------------------------------------------
// Host orchestration
// ---------------------------------------------------------------------------
static const int Bc = 32, Nn = 512, Tt = 20, VMAX = 96, PDMAX = 64, NEc = 16;
static const int ENC = 128, ATOT = VMAX + PDMAX; // 160

struct Bump {
    char* p;
    void* take(size_t bytes) {
        char* r = p;
        p += (bytes + 255) & ~(size_t)255;
        return (void*)r;
    }
};

static void gemm(hipStream_t s, const __bf16* A, const __bf16* W, const float* bias,
                 float* C, __bf16* Cbf, int M, int N, int K, int act) {
    dim3 g((M + 15) / 16, N / 64);
    wmma_gemm_k<<<g, 32, 0, s>>>(A, W, bias, C, Cbf, M, N, K, act);
}

extern "C" void kernel_launch(void* const* d_in, const int* in_sizes, int n_in,
                              void* d_out, int out_size, void* d_ws, size_t ws_size,
                              hipStream_t stream) {
    (void)in_sizes; (void)n_in; (void)out_size; (void)ws_size;
    // ---- input indices (setup_inputs order; params pytree-flattened with
    //      alphabetical dict keys at every level) ----
    const float* taf = (const float*)d_in[0];
    const float* lnf = (const float*)d_in[1];
    const float* lnm = (const float*)d_in[2];
    const float* nvf = (const float*)d_in[3];
    const float* nvm = (const float*)d_in[4];
    const float* npf = (const float*)d_in[5];
    const float* npm = (const float*)d_in[6];
    const unsigned char* vnm = (const unsigned char*)d_in[7];
    const unsigned char* pnm = (const unsigned char*)d_in[8];
    const int* s_next = (const int*)d_in[9];
    const int* edge_t = (const int*)d_in[10];

    Bump ws{ (char*)d_ws };
    auto cvt = [&](int idx, size_t n) -> __bf16* {
        __bf16* dst = (__bf16*)ws.take(n * sizeof(__bf16));
        f32_to_bf16_k<<<dim3((unsigned)((n + 255) / 256)), 256, 0, stream>>>((const float*)d_in[idx], dst, (int)n);
        return dst;
    };

    // ---- parameter conversion to bf16 ----
    __bf16* anW_in  = cvt(11, 384 * 128);
    __bf16* anW_out = cvt(13, 128 * 128);
    __bf16 *gatt_inW[3], *gatt_outW[3], *gkW[3], *gqW[3], *gvW[3];
    const float *gatt_inb[3], *gatt_outb[3], *gkb[3], *gqb[3], *gvb[3];
    for (int l = 0; l < 3; ++l) {
        int base = 15 + 10 * l;
        gatt_inW[l]  = cvt(base + 0, 384 * 128);
        gatt_inb[l]  = (const float*)d_in[base + 1];
        gatt_outW[l] = cvt(base + 2, 128 * 128);
        gatt_outb[l] = (const float*)d_in[base + 3];
        gkW[l] = cvt(base + 4, 128 * 128); gkb[l] = (const float*)d_in[base + 5];
        gqW[l] = cvt(base + 6, 128 * 128); gqb[l] = (const float*)d_in[base + 7];
        gvW[l] = cvt(base + 8, 128 * 128); gvb[l] = (const float*)d_in[base + 9];
    }
    __bf16* kembW = cvt(45, 128 * 128);
    __bf16* mixW  = cvt(47, 128 * 256);
    __bf16* nbrWhh = cvt(51, 384 * 128);
    __bf16* nbrWih = cvt(52, 384 * 64);
    __bf16* nodeWhh = cvt(57, 384 * 128);
    __bf16* nodeWih = cvt(58, 384 * 64);
    __bf16* qembW = cvt(61, 128 * 128);
    __bf16* taWhh = cvt(65, 384 * 128);
    __bf16* taWih = cvt(66, 384 * 64);
    __bf16* vembW = cvt(69, 128 * 128);

    // ---- workspace activations ----
    const int MN = Bc * Nn;          // 16384 node sequences
    const int MR = Bc * ATOT;        // 5120 neighbor sequences
    __bf16* ebT = (__bf16*)ws.take((size_t)Bc * Tt * 64 * 2);
    __bf16* ebN = (__bf16*)ws.take((size_t)MN * Tt * 64 * 2);
    __bf16* ebR = (__bf16*)ws.take((size_t)MR * Tt * 64 * 2);
    unsigned char* validT = (unsigned char*)ws.take((size_t)Bc * Tt);
    unsigned char* validN = (unsigned char*)ws.take((size_t)MN * Tt);
    unsigned char* validR = (unsigned char*)ws.take((size_t)MR * Tt);
    float*  hN   = (float*)ws.take((size_t)MN * 128 * 4);
    __bf16* hNbf = (__bf16*)ws.take((size_t)MN * 128 * 2);
    float*  hR   = (float*)ws.take((size_t)MR * 128 * 4);
    __bf16* hRbf = (__bf16*)ws.take((size_t)MR * 128 * 2);
    float*  dump = (float*)ws.take((size_t)MN * 128 * 4);   // shared f32 sink
    __bf16* qbf  = (__bf16*)ws.take((size_t)MN * 128 * 2);
    __bf16* kbf  = (__bf16*)ws.take((size_t)MR * 128 * 2);
    __bf16* vbf  = (__bf16*)ws.take((size_t)MR * 128 * 2);
    __bf16* qhbf = (__bf16*)ws.take((size_t)MN * 128 * 2);
    __bf16* khbf = (__bf16*)ws.take((size_t)MR * 128 * 2);
    __bf16* vhbf = (__bf16*)ws.take((size_t)MR * 128 * 2);
    float*  ctx   = (float*)ws.take((size_t)MN * 128 * 4);
    __bf16* ctxbf = (__bf16*)ws.take((size_t)MN * 128 * 2);
    __bf16* attpbf = (__bf16*)ws.take((size_t)MN * 128 * 2);
    __bf16* mixin  = (__bf16*)ws.take((size_t)MN * 256 * 2);
    float*  ne    = (float*)ws.take((size_t)MN * 128 * 4);
    __bf16* nebf  = (__bf16*)ws.take((size_t)MN * 128 * 2);
    __bf16* t1 = (__bf16*)ws.take((size_t)MN * 128 * 2);
    __bf16* t2 = (__bf16*)ws.take((size_t)MN * 128 * 2);
    __bf16* t3 = (__bf16*)ws.take((size_t)MN * 128 * 2);
    __bf16* t4 = (__bf16*)ws.take((size_t)MN * 128 * 2);
    __bf16* t5 = (__bf16*)ws.take((size_t)MN * 128 * 2);
    __bf16* t6 = (__bf16*)ws.take((size_t)MN * 128 * 2);
    float*  proj = (float*)ws.take((size_t)MN * 128 * 4);
    unsigned char* attmask = (unsigned char*)ws.take((size_t)Bc * Nn * ATOT);
    unsigned char* adjmask = (unsigned char*)ws.take((size_t)Bc * Nn * Nn);

    // ---- embeddings (and validity) ----
    {
        int tot = Bc * 1 * Tt * 64;
        embed_k<<<dim3((tot + 255) / 256), 256, 0, stream>>>(taf, (const float*)nullptr,
            (const float*)d_in[63], (const float*)d_in[64], ebT, validT,
            Bc, 1, 1, 0, Tt, 8, 8, -1.f);
        tot = Bc * Nn * Tt * 64;
        embed_k<<<dim3((tot + 255) / 256), 256, 0, stream>>>(lnf, lnm,
            (const float*)d_in[55], (const float*)d_in[56], ebN, validN,
            Bc, Nn, Nn, 0, Tt, 8, 8, -1.f);
        tot = Bc * VMAX * Tt * 64;
        embed_k<<<dim3((tot + 255) / 256), 256, 0, stream>>>(nvf, nvm,
            (const float*)d_in[49], (const float*)d_in[50], ebR, validR,
            Bc, VMAX, ATOT, 0, Tt, 8, 9, 0.f);
        tot = Bc * PDMAX * Tt * 64;
        embed_k<<<dim3((tot + 255) / 256), 256, 0, stream>>>(npf, npm,
            (const float*)d_in[49], (const float*)d_in[50], ebR, validR,
            Bc, PDMAX, ATOT, VMAX, Tt, 8, 9, 1.f);
    }

    // ---- GRU encoders (WMMA recurrences) ----
    float* ta_out = (float*)d_out; // [32,128] first output
    gru_fused_k<<<dim3(Bc / 16), 32, 0, stream>>>(ebT, validT, taWih, taWhh,
        (const float*)d_in[68], (const float*)d_in[67], ta_out, (__bf16*)nullptr, Bc, Tt);
    gru_fused_k<<<dim3(MN / 16), 32, 0, stream>>>(ebN, validN, nodeWih, nodeWhh,
        (const float*)d_in[60], (const float*)d_in[59], hN, hNbf, MN, Tt);
    gru_fused_k<<<dim3(MR / 16), 32, 0, stream>>>(ebR, validR, nbrWih, nbrWhh,
        (const float*)d_in[54], (const float*)d_in[53], hR, hRbf, MR, Tt);

    // ---- attention masks ----
    {
        int tot = Bc * Nn * ATOT;
        build_attmask_k<<<dim3((tot + 255) / 256), 256, 0, stream>>>(vnm, pnm, attmask, tot);
        build_adjmask_k<<<dim3(Bc * Nn), 128, 0, stream>>>(s_next, edge_t, adjmask, Nn, NEc);
    }

    // ---- agent-node cross attention ----
    const float scale = 1.f / sqrtf((float)ENC);
    gemm(stream, hNbf, qembW, (const float*)d_in[62], dump, qbf, MN, 128, 128, 0);
    gemm(stream, hRbf, kembW, (const float*)d_in[46], dump, kbf, MR, 128, 128, 0);
    gemm(stream, hRbf, vembW, (const float*)d_in[70], dump, vbf, MR, 128, 128, 0);
    gemm(stream, qbf, anW_in + 0 * 128 * 128, (const float*)d_in[12] + 0,   dump, qhbf, MN, 128, 128, 0);
    gemm(stream, kbf, anW_in + 1 * 128 * 128, (const float*)d_in[12] + 128, dump, khbf, MR, 128, 128, 0);
    gemm(stream, vbf, anW_in + 2 * 128 * 128, (const float*)d_in[12] + 256, dump, vhbf, MR, 128, 128, 0);
    attn_k<<<dim3(Nn / 16, Bc), 32, 0, stream>>>(qhbf, khbf, vhbf, attmask, ctx, ctxbf, Nn, ATOT, scale);
    gemm(stream, ctxbf, anW_out, (const float*)d_in[14], dump, attpbf, MN, 128, 128, 0);
    {
        int tot = MN * 256;
        concat2_bf_k<<<dim3((tot + 255) / 256), 256, 0, stream>>>(hNbf, attpbf, mixin, MN);
    }
    gemm(stream, mixin, mixW, (const float*)d_in[48], ne, nebf, MN, 128, 256, 1);

    // ---- GAT layers with residuals ----
    for (int l = 0; l < 3; ++l) {
        gemm(stream, nebf, gqW[l], gqb[l], dump, t1, MN, 128, 128, 0);
        gemm(stream, nebf, gkW[l], gkb[l], dump, t2, MN, 128, 128, 0);
        gemm(stream, nebf, gvW[l], gvb[l], dump, t3, MN, 128, 128, 0);
        gemm(stream, t1, gatt_inW[l] + 0 * 128 * 128, gatt_inb[l] + 0,   dump, t4, MN, 128, 128, 0);
        gemm(stream, t2, gatt_inW[l] + 1 * 128 * 128, gatt_inb[l] + 128, dump, t5, MN, 128, 128, 0);
        gemm(stream, t3, gatt_inW[l] + 2 * 128 * 128, gatt_inb[l] + 256, dump, t6, MN, 128, 128, 0);
        attn_k<<<dim3(Nn / 16, Bc), 32, 0, stream>>>(t4, t5, t6, adjmask, ctx, ctxbf, Nn, Nn, scale);
        gemm(stream, ctxbf, gatt_outW[l], gatt_outb[l], proj, (__bf16*)nullptr, MN, 128, 128, 0);
        int tot = MN * 128;
        add_resid_k<<<dim3((tot + 255) / 256), 256, 0, stream>>>(ne, proj, nebf, tot);
    }

    // ---- final outputs: [ta_enc | node_enc | combined_masks] ----
    float* node_out = (float*)d_out + (size_t)Bc * 128;
    float* mask_out = node_out + (size_t)MN * 128;
    {
        int tot = MN * 128;
        copy_f32_k<<<dim3((tot + 255) / 256), 256, 0, stream>>>(ne, node_out, tot);
        combined_mask_k<<<dim3((MN + 255) / 256), 256, 0, stream>>>(validN, mask_out, MN, Tt);
    }
}
